// ModelA_30245159698598
// MI455X (gfx1250) — compile-verified, matrix operand order checked by reading, not run on hardware
//
#include <hip/hip_runtime.h>
#include <math.h>

typedef __attribute__((ext_vector_type(16))) __bf16 v16bf;
typedef __attribute__((ext_vector_type(8)))  float  v8f;

#define NEGV (-1e30f)

// ---------- helpers ----------
__device__ inline __bf16 f2bf(float f) {
  return static_cast<__bf16>(f);   // hardware RNE convert on gfx1250
}

__device__ inline v16bf zero_frag() {
  v16bf r; __bf16 z = f2bf(0.f);
#pragma unroll
  for (int i = 0; i < 16; i++) r[i] = z;
  return r;
}

__device__ inline v16bf load_frag_contig(const float* __restrict__ p) {
  v16bf r;
#pragma unroll
  for (int i = 0; i < 8; i++) r[i] = f2bf(p[i]);
#pragma unroll
  for (int i = 0; i < 8; i++) r[i + 8] = f2bf(p[16 + i]);
  return r;
}
__device__ inline v16bf load_frag_strided(const float* __restrict__ p, int ld) {
  v16bf r;
#pragma unroll
  for (int i = 0; i < 8; i++) r[i] = f2bf(p[(size_t)i * ld]);
#pragma unroll
  for (int i = 0; i < 8; i++) r[i + 8] = f2bf(p[(size_t)(16 + i) * ld]);
  return r;
}

// ---------- WMMA GEMM ----------
// C[M,N] = A * B (+bias[n]).  AT=0: A[m*lda+k] ; AT=1: A[k*lda+m]
// BT=0: B is weights W[n*ldb+k] ; BT=1: B[k*ldb+n]
// Batched: z = z1*innerB + z0 ; ptr += z1*S1 + z0*S0
// Requires: M%16==0, N%16==0, K%32==0.
// One wave computes a 32x64 C tile (2 M-subtiles x 4 N-subtiles, A frags
// reused across the 4 B frags); 4 waves/block -> 128x64 per block.
// Wave id is pushed through readfirstlane so the m-subtile guards are
// provably scalar (s_cbranch, never EXEC-masked WMMA — ISA requires EXEC
// all-ones around WMMA).
template <bool AT, bool BT>
__global__ __launch_bounds__(128) void gemm_wmma(
    const float* __restrict__ A, const float* __restrict__ B,
    float* __restrict__ C, const float* __restrict__ bias,
    int M, int N, int K, int lda, int ldb, int ldc,
    long long aS0, long long aS1, long long bS0, long long bS1,
    long long cS0, long long cS1, int innerB)
{
  int z  = blockIdx.z;
  int z1 = z / innerB, z0 = z % innerB;
  A += z1 * aS1 + z0 * aS0;
  B += z1 * bS1 + z0 * bS0;
  C += z1 * cS1 + z0 * cS0;

  int wave = __builtin_amdgcn_readfirstlane((int)threadIdx.x >> 5);  // SGPR
  int lane = threadIdx.x & 31;
  int m0 = blockIdx.y * 128 + wave * 32;     // scalar
  if (m0 >= M) return;                       // scalar branch
  bool m1ok = (m0 + 16) < M;                 // scalar
  int n0 = blockIdx.x * 64;
  int r    = lane & 15;
  int koff = (lane >> 4) * 8;

  v8f acc[2][4] = {};
  for (int k = 0; k < K; k += 32) {
    // prefetch next K tile (lowers to global_prefetch_b8)
    if (k + 32 < K) {
      if (!AT) __builtin_prefetch(A + (size_t)(m0 + r) * lda + (k + 32) + koff, 0, 1);
      else     __builtin_prefetch(A + (size_t)(k + 32 + koff) * lda + (m0 + r), 0, 1);
      if (!BT) __builtin_prefetch(B + (size_t)(n0 + r) * ldb + (k + 32) + koff, 0, 1);
      else     __builtin_prefetch(B + (size_t)(k + 32 + koff) * ldb + (n0 + r), 0, 1);
    }
    v16bf a0, a1 = zero_frag();
    if (!AT) {
      const float* pa = A + (size_t)(m0 + r) * lda + k + koff;
      a0 = load_frag_contig(pa);
      if (m1ok) a1 = load_frag_contig(pa + (size_t)16 * lda);
    } else {
      const float* pa = A + (size_t)(k + koff) * lda + (m0 + r);
      a0 = load_frag_strided(pa, lda);
      if (m1ok) a1 = load_frag_strided(pa + 16, lda);
    }
#pragma unroll
    for (int t = 0; t < 4; t++) {
      int n0t = n0 + t * 16;
      if (n0t >= N) break;                   // scalar (blockIdx-based)
      v16bf b;
      if (!BT) b = load_frag_contig(B + (size_t)(n0t + r) * ldb + k + koff);
      else     b = load_frag_strided(B + (size_t)(k + koff) * ldb + (n0t + r), ldb);
      acc[0][t] = __builtin_amdgcn_wmma_f32_16x16x32_bf16(
          false, a0, false, b, (short)0, acc[0][t], false, false);
      if (m1ok)
        acc[1][t] = __builtin_amdgcn_wmma_f32_16x16x32_bf16(
            false, a1, false, b, (short)0, acc[1][t], false, false);
    }
  }
  int hi = (lane >> 4) * 8;
#pragma unroll
  for (int s = 0; s < 2; s++) {
    if (s == 1 && !m1ok) break;
#pragma unroll
    for (int t = 0; t < 4; t++) {
      int n0t = n0 + t * 16;
      if (n0t >= N) break;
      int n = n0t + r;
      float bv = bias ? bias[n] : 0.0f;
#pragma unroll
      for (int v = 0; v < 8; v++) {
        int m = m0 + s * 16 + hi + v;
        C[(size_t)m * ldc + n] = acc[s][t][v] + bv;
      }
    }
  }
}

// ---------- elementwise / reduction kernels ----------
__global__ void transpose_k(const float* __restrict__ x, float* __restrict__ y,
                            int D, int L, int n) {
  int idx = blockIdx.x * blockDim.x + threadIdx.x;
  if (idx >= n) return;
  int d = idx % D; size_t row = (size_t)idx / D; int l = (int)(row % L); int b = (int)(row / L);
  y[idx] = x[((size_t)b * D + d) * L + l];
}

__global__ void dwconv_k(const float* __restrict__ x, const float* __restrict__ dw,
                         const float* __restrict__ dwb, float* __restrict__ y,
                         int L, int D, int K, int n) {
  int idx = blockIdx.x * blockDim.x + threadIdx.x;
  if (idx >= n) return;
  int d = idx % D; size_t row = (size_t)idx / D; int l = (int)(row % L); int b = (int)(row / L);
  const float* xb = x + (size_t)b * L * D + d;
  float acc = dwb[d]; int p0 = K / 2;
  for (int j = 0; j < K; j++) {
    int pos = l + j - p0;
    if (pos >= 0 && pos < L) acc += xb[(size_t)pos * D] * dw[d * K + j];
  }
  y[idx] = acc;
}

__global__ void add_pe_k(const float* __restrict__ x, const float* __restrict__ pe,
                         float* __restrict__ y, int L, int D, int n) {
  int idx = blockIdx.x * blockDim.x + threadIdx.x;
  if (idx >= n) return;
  int d = idx % D; int l = (int)(((size_t)idx / D) % L);
  y[idx] = x[idx] + pe[(size_t)d * L + l];
}

__global__ void relu_add_k(const float* __restrict__ x, const float* __restrict__ r,
                           float* __restrict__ y, int n) {
  int idx = blockIdx.x * blockDim.x + threadIdx.x;
  if (idx >= n) return;
  y[idx] = fmaxf(x[idx], 0.0f) + r[idx];
}

__global__ void add_k(const float* __restrict__ x, const float* __restrict__ r,
                      float* __restrict__ y, int n) {
  int idx = blockIdx.x * blockDim.x + threadIdx.x;
  if (idx >= n) return;
  y[idx] = x[idx] + r[idx];
}

__global__ void mul_vec_k(const float* __restrict__ x, const float* __restrict__ w,
                          float* __restrict__ y, int D, int n) {
  int idx = blockIdx.x * blockDim.x + threadIdx.x;
  if (idx >= n) return;
  y[idx] = x[idx] * w[idx % D];
}

__global__ __launch_bounds__(256) void ln_stats_k(const float* __restrict__ x,
                                                  float* __restrict__ stats, int ne) {
  __shared__ float s1[256], s2[256];
  int b = blockIdx.x;
  const float* p = x + (size_t)b * ne;
  float su = 0.f, sq = 0.f;
  for (int i = threadIdx.x; i < ne; i += 256) { float v = p[i]; su += v; sq += v * v; }
  s1[threadIdx.x] = su; s2[threadIdx.x] = sq; __syncthreads();
  for (int o = 128; o > 0; o >>= 1) {
    if ((int)threadIdx.x < o) { s1[threadIdx.x] += s1[threadIdx.x + o]; s2[threadIdx.x] += s2[threadIdx.x + o]; }
    __syncthreads();
  }
  if (threadIdx.x == 0) {
    float mu = s1[0] / (float)ne;
    float var = s2[0] / (float)ne - mu * mu;
    stats[b * 2] = mu;
    stats[b * 2 + 1] = rsqrtf(fmaxf(var, 0.f) + 1e-5f);
  }
}

__global__ void ln_apply_k(const float* __restrict__ x, const float* __restrict__ stats,
                           const float* __restrict__ g, const float* __restrict__ bb,
                           float* __restrict__ y, int L, int D, int n) {
  int idx = blockIdx.x * blockDim.x + threadIdx.x;
  if (idx >= n) return;
  int d = idx % D; size_t row = (size_t)idx / D; int l = (int)(row % L); int b = (int)(row / L);
  float mu = stats[b * 2], rs = stats[b * 2 + 1];
  size_t gi = (size_t)d * L + l;
  y[idx] = (x[idx] - mu) * rs * g[gi] + bb[gi];
}

__global__ __launch_bounds__(256) void softmax_rows_k(float* __restrict__ S,
                                                      const float* __restrict__ mask,
                                                      int cols, int rowsPerBatch, float scale) {
  __shared__ float sh[256];
  int row = blockIdx.x;
  int b = row / rowsPerBatch;
  float* s = S + (size_t)row * cols;
  const float* m = mask + (size_t)b * cols;
  int t = threadIdx.x;
  float mx = -3.4e38f;
  for (int j = t; j < cols; j += 256) {
    float mm = m[j];
    float v = s[j] * scale * mm + (1.0f - mm) * NEGV;
    s[j] = v; mx = fmaxf(mx, v);
  }
  sh[t] = mx; __syncthreads();
  for (int o = 128; o > 0; o >>= 1) { if (t < o) sh[t] = fmaxf(sh[t], sh[t + o]); __syncthreads(); }
  mx = sh[0]; __syncthreads();
  float sum = 0.f;
  for (int j = t; j < cols; j += 256) { float e = __expf(s[j] - mx); s[j] = e; sum += e; }
  sh[t] = sum; __syncthreads();
  for (int o = 128; o > 0; o >>= 1) { if (t < o) sh[t] += sh[t + o]; __syncthreads(); }
  float inv = 1.0f / sh[0];
  for (int j = t; j < cols; j += 256) s[j] *= inv;
}

__global__ void softmax_cols_k(float* __restrict__ S, const float* __restrict__ mask,
                               int N, int Mc, int Btot) {
  int idx = blockIdx.x * blockDim.x + threadIdx.x;
  if (idx >= Btot * Mc) return;
  int b = idx / Mc, m = idx % Mc;
  float* base = S + (size_t)b * N * Mc + m;
  const float* mk = mask + (size_t)b * N;
  float mx = -3.4e38f;
  for (int n = 0; n < N; n++) {
    float mm = mk[n];
    float v = base[(size_t)n * Mc] * mm + (1.f - mm) * NEGV;
    base[(size_t)n * Mc] = v; mx = fmaxf(mx, v);
  }
  float sum = 0.f;
  for (int n = 0; n < N; n++) { float e = __expf(base[(size_t)n * Mc] - mx); base[(size_t)n * Mc] = e; sum += e; }
  float inv = 1.f / sum;
  for (int n = 0; n < N; n++) base[(size_t)n * Mc] *= inv;
}

__global__ void cqa_bias_k(float* __restrict__ S, const float* __restrict__ c1,
                           const float* __restrict__ q2, int N, int Mc, int n) {
  int idx = blockIdx.x * blockDim.x + threadIdx.x;
  if (idx >= n) return;
  int m = idx % Mc; size_t row = (size_t)idx / Mc; int nn = (int)(row % N); int b = (int)(row / N);
  S[idx] += c1[(size_t)b * N + nn] + q2[(size_t)b * Mc + m];
}

__global__ void cqa_concat_k(const float* __restrict__ Ct, const float* __restrict__ A,
                             const float* __restrict__ Bm, float* __restrict__ X, int n) {
  int idx = blockIdx.x * blockDim.x + threadIdx.x;
  if (idx >= n) return;
  int d = idx & 1023; size_t row = (size_t)idx >> 10;
  int which = d >> 8; int dd = d & 255;
  size_t ri = row * 256 + dd;
  float c = Ct[ri]; float v;
  if (which == 0) v = c;
  else if (which == 1) v = A[ri];
  else if (which == 2) v = c * A[ri];
  else v = c * Bm[ri];
  X[idx] = v;
}

__global__ void dot2_k(const float* __restrict__ A, const float* __restrict__ B2,
                       const float* __restrict__ w, float* __restrict__ y, int rows, int D) {
  int idx = blockIdx.x * blockDim.x + threadIdx.x;
  if (idx >= rows) return;
  const float* a = A + (size_t)idx * D;
  float s = 0.f;
  for (int d = 0; d < D; d++) s += a[d] * w[d];
  if (B2) {
    const float* b = B2 + (size_t)idx * D;
    for (int d = 0; d < D; d++) s += b[d] * w[D + d];
  }
  y[idx] = s;
}

// ---------- host ----------
extern "C" void kernel_launch(void* const* d_in, const int* in_sizes, int n_in,
                              void* d_out, int out_size, void* d_ws, size_t ws_size,
                              hipStream_t stream) {
  (void)in_sizes; (void)n_in; (void)out_size; (void)ws_size;
  const int Bn = 16, LC = 512, LQ = 64, DW = 768, DM = 256, NH = 8, DK = 32;

  // Param leaf indices (JAX pytree flatten: dicts sorted by key, lists in order)
  enum {
    IN_CTX = 0, IN_Q = 1, IN_CMASK = 2, IN_QMASK = 3,
    P_CC_DW = 4, P_CC_DWB = 5, P_CC_PW = 6, P_CC_PWB = 7,
    P_CQA_W = 8,
    P_ENC_C = 9,            // 39 leaves (cn=4)
    P_ENC_Q = 48,           // 39 leaves
    P_PTR_L1B = 87, P_PTR_L1W = 88, P_PTR_L2B = 89, P_PTR_L2W = 90,
    P_PTR_W1 = 91, P_PTR_W2 = 92,
    P_QC_DW = 93, P_QC_DWB = 94, P_QC_PW = 95, P_QC_PWB = 96,
    P_RS_DW = 97, P_RS_DWB = 98, P_RS_PW = 99, P_RS_PWB = 100,
    P_STK = 101             // 27 leaves (cn=2)
  };
  auto in = [&](int i) { return (const float*)d_in[i]; };
  const float* cmask = in(IN_CMASK);
  const float* qmask = in(IN_QMASK);

  float* ws = (float*)d_ws;
  size_t off = 0;
  auto alloc = [&](size_t n) { float* p = ws + off; off += n; return p; };
  const size_t NBL = (size_t)Bn * LC * DM;

  float* T0   = alloc((size_t)Bn * LC * 1024);
  float* T1   = alloc((size_t)Bn * LC * 1024);
  float* ATTN = alloc((size_t)Bn * NH * LC * LC);
  float* EB_T = alloc(NBL); float* EB_U = alloc(NBL); float* EB_T2 = alloc(NBL);
  float* EB_Q = alloc(NBL); float* EB_K = alloc(NBL); float* EB_V  = alloc(NBL);
  float* EB_R0 = alloc(NBL); float* EB_R1 = alloc(NBL);
  float* Cbuf = alloc(NBL); float* Cebuf = alloc(NBL);
  float* Qbuf = alloc(NBL); float* Qebuf = alloc(NBL);
  float* Ctw3 = alloc(NBL); float* Acqa = alloc(NBL); float* Bmb = alloc(NBL);
  float* M1b = alloc(NBL); float* M2b = alloc(NBL); float* M3b = alloc(NBL);
  float* Sb  = alloc((size_t)Bn * LC * LQ);
  float* S1b = alloc((size_t)Bn * LC * LQ);
  float* S2b = alloc((size_t)Bn * LC * LQ);
  float* Ub  = alloc((size_t)Bn * LQ * DM);
  float* C1v = alloc((size_t)Bn * LC);
  float* Q2v = alloc((size_t)Bn * LQ);
  float* Y1v = alloc((size_t)Bn * LC);
  float* Y2v = alloc((size_t)Bn * LC);
  float* STATS = alloc(64);

  auto grid1 = [](size_t n) { return dim3((unsigned)((n + 255) / 256)); };

  auto gemm = [&](bool at, bool bt, const float* A, const float* B, float* C,
                  const float* bias, int M, int N, int K, int lda, int ldb, int ldc,
                  int batches, int innerB,
                  long long aS0, long long aS1, long long bS0, long long bS1,
                  long long cS0, long long cS1) {
    dim3 g((unsigned)((N + 63) / 64), (unsigned)((M + 127) / 128), (unsigned)batches);
    dim3 blk(128);
    if (!at && !bt)
      gemm_wmma<false, false><<<g, blk, 0, stream>>>(A, B, C, bias, M, N, K, lda, ldb, ldc, aS0, aS1, bS0, bS1, cS0, cS1, innerB);
    else if (!at && bt)
      gemm_wmma<false, true><<<g, blk, 0, stream>>>(A, B, C, bias, M, N, K, lda, ldb, ldc, aS0, aS1, bS0, bS1, cS0, cS1, innerB);
    else if (at && bt)
      gemm_wmma<true, true><<<g, blk, 0, stream>>>(A, B, C, bias, M, N, K, lda, ldb, ldc, aS0, aS1, bS0, bS1, cS0, cS1, innerB);
    else
      gemm_wmma<true, false><<<g, blk, 0, stream>>>(A, B, C, bias, M, N, K, lda, ldb, ldc, aS0, aS1, bS0, bS1, cS0, cS1, innerB);
  };
  auto gemmS = [&](const float* A, const float* B, float* C, const float* bias,
                   int M, int N, int K, int lda, int ldb, int ldc) {
    gemm(false, false, A, B, C, bias, M, N, K, lda, ldb, ldc, 1, 1, 0, 0, 0, 0, 0, 0);
  };

  auto dsconv = [&](const float* x, int L, int Cin, int Cout, int k,
                    int dwI, int dwbI, int pwI, int pwbI, float* tmp, float* out) {
    size_t n = (size_t)Bn * L * Cin;
    dwconv_k<<<grid1(n), 256, 0, stream>>>(x, in(dwI), in(dwbI), tmp, L, Cin, k, (int)n);
    gemmS(tmp, in(pwI), out, in(pwbI), Bn * L, Cout, Cin, Cin, Cin, Cout);
  };

  auto encoder = [&](const float* x, float* out, int base, int cn, int L,
                     const float* mask, int kconv) {
    int fcb = base + 4 * cn, fcw = fcb + 1;
    int mfcb = fcb + 2, mfcw = fcb + 3, mkb = fcb + 4, mkw = fcb + 5;
    int mqb = fcb + 6, mqw = fcb + 7, mvb = fcb + 8, mvw = fcb + 9;
    int nbb = fcb + 10, nbg = fcb + 11, neb = fcb + 12, negI = fcb + 13;
    int ns0 = fcb + 14;
    int peI = ns0 + 2 * cn;
    size_t n = (size_t)Bn * L * DM;
    int ne = DM * L;

    add_pe_k<<<grid1(n), 256, 0, stream>>>(x, in(peI), EB_R0, L, DM, (int)n);
    const float* res = EB_R0; float* resN = EB_R1;
    ln_stats_k<<<dim3(Bn), 256, 0, stream>>>(res, STATS, ne);
    ln_apply_k<<<grid1(n), 256, 0, stream>>>(res, STATS, in(nbg), in(nbb), EB_T, L, DM, (int)n);

    for (int i = 0; i < cn; i++) {
      dsconv(EB_T, L, DM, DM, kconv, base + 4 * i, base + 4 * i + 1,
             base + 4 * i + 2, base + 4 * i + 3, EB_U, EB_T2);
      relu_add_k<<<grid1(n), 256, 0, stream>>>(EB_T2, res, resN, (int)n);
      res = resN; resN = (res == EB_R1) ? EB_R0 : EB_R1;
      ln_stats_k<<<dim3(Bn), 256, 0, stream>>>(res, STATS, ne);
      ln_apply_k<<<grid1(n), 256, 0, stream>>>(res, STATS, in(ns0 + 2 * i + 1), in(ns0 + 2 * i), EB_T, L, DM, (int)n);
    }

    // MHA on EB_T
    gemmS(EB_T, in(mqw), EB_Q, in(mqb), Bn * L, DM, DM, DM, DM, DM);
    gemmS(EB_T, in(mkw), EB_K, in(mkb), Bn * L, DM, DM, DM, DM, DM);
    gemmS(EB_T, in(mvw), EB_V, in(mvb), Bn * L, DM, DM, DM, DM, DM);
    // scores: per (b,h): [L,L] = Q[L,32] x K^T
    gemm(false, false, EB_Q, EB_K, ATTN, nullptr, L, L, DK, DM, DM, L,
         Bn * NH, NH, DK, (long long)L * DM, DK, (long long)L * DM,
         (long long)L * L, (long long)NH * L * L);
    softmax_rows_k<<<dim3((unsigned)(Bn * NH * L)), 256, 0, stream>>>(ATTN, mask, L, NH * L, 0.17677669529663687f);
    // context: per (b,h): [L,32] = attn[L,L] x V (V is K-major)
    gemm(false, true, ATTN, EB_V, EB_U, nullptr, L, DK, L, L, DM, DM,
         Bn * NH, NH, (long long)L * L, (long long)NH * L * L,
         DK, (long long)L * DM, DK, (long long)L * DM);
    gemmS(EB_U, in(mfcw), EB_T2, in(mfcb), Bn * L, DM, DM, DM, DM, DM);
    add_k<<<grid1(n), 256, 0, stream>>>(EB_T2, res, resN, (int)n);
    res = resN; resN = (res == EB_R1) ? EB_R0 : EB_R1;

    ln_stats_k<<<dim3(Bn), 256, 0, stream>>>(res, STATS, ne);
    ln_apply_k<<<grid1(n), 256, 0, stream>>>(res, STATS, in(negI), in(neb), EB_T, L, DM, (int)n);
    gemmS(EB_T, in(fcw), EB_T2, in(fcb), Bn * L, DM, DM, DM, DM, DM);
    relu_add_k<<<grid1(n), 256, 0, stream>>>(EB_T2, res, out, (int)n);
  };

  // ---- input convs ----
  {
    size_t nct = (size_t)Bn * LC * DW;
    transpose_k<<<grid1(nct), 256, 0, stream>>>(in(IN_CTX), T0, DW, LC, (int)nct);
    dsconv(T0, LC, DW, DM, 5, P_CC_DW, P_CC_DWB, P_CC_PW, P_CC_PWB, T1, Cbuf);
    size_t nqt = (size_t)Bn * LQ * DW;
    transpose_k<<<grid1(nqt), 256, 0, stream>>>(in(IN_Q), T0, DW, LQ, (int)nqt);
    dsconv(T0, LQ, DW, DM, 5, P_QC_DW, P_QC_DWB, P_QC_PW, P_QC_PWB, T1, Qbuf);
  }

  // ---- embedding encoders ----
  encoder(Cbuf, Cebuf, P_ENC_C, 4, LC, cmask, 7);
  encoder(Qbuf, Qebuf, P_ENC_Q, 4, LQ, qmask, 7);

  // ---- context-query attention ----
  mul_vec_k<<<grid1(NBL), 256, 0, stream>>>(Cebuf, in(P_CQA_W) + 2 * DM, Ctw3, DM, (int)NBL);
  dot2_k<<<grid1((size_t)Bn * LC), 256, 0, stream>>>(Cebuf, nullptr, in(P_CQA_W), C1v, Bn * LC, DM);
  dot2_k<<<grid1((size_t)Bn * LQ), 256, 0, stream>>>(Qebuf, nullptr, in(P_CQA_W) + DM, Q2v, Bn * LQ, DM);
  gemm(false, false, Ctw3, Qebuf, Sb, nullptr, LC, LQ, DM, DM, DM, LQ,
       Bn, 1, 0, (long long)LC * DM, 0, (long long)LQ * DM, 0, (long long)LC * LQ);
  cqa_bias_k<<<grid1((size_t)Bn * LC * LQ), 256, 0, stream>>>(Sb, C1v, Q2v, LC, LQ, Bn * LC * LQ);
  hipMemcpyAsync(S1b, Sb, sizeof(float) * (size_t)Bn * LC * LQ, hipMemcpyDeviceToDevice, stream);
  hipMemcpyAsync(S2b, Sb, sizeof(float) * (size_t)Bn * LC * LQ, hipMemcpyDeviceToDevice, stream);
  softmax_rows_k<<<dim3((unsigned)(Bn * LC)), 256, 0, stream>>>(S1b, qmask, LQ, LC, 1.0f);
  softmax_cols_k<<<grid1((size_t)Bn * LQ), 256, 0, stream>>>(S2b, cmask, LC, LQ, Bn);
  // A = S1 @ Qt   (Qt is K-major)
  gemm(false, true, S1b, Qebuf, Acqa, nullptr, LC, DM, LQ, LQ, DM, DM,
       Bn, 1, 0, (long long)LC * LQ, 0, (long long)LQ * DM, 0, (long long)LC * DM);
  // U = S2^T @ Ct
  gemm(true, true, S2b, Cebuf, Ub, nullptr, LQ, DM, LC, LQ, DM, DM,
       Bn, 1, 0, (long long)LC * LQ, 0, (long long)LC * DM, 0, (long long)LQ * DM);
  // Bm = S1 @ U
  gemm(false, true, S1b, Ub, Bmb, nullptr, LC, DM, LQ, LQ, DM, DM,
       Bn, 1, 0, (long long)LC * LQ, 0, (long long)LQ * DM, 0, (long long)LC * DM);
  cqa_concat_k<<<grid1((size_t)Bn * LC * 1024), 256, 0, stream>>>(Cebuf, Acqa, Bmb, T0, (int)((size_t)Bn * LC * 1024));

  // ---- resizer ----
  dsconv(T0, LC, 1024, DM, 5, P_RS_DW, P_RS_DWB, P_RS_PW, P_RS_PWB, T1, M1b);

  // ---- stacked model encoders (in-place safe: input read first, output written last) ----
  for (int i = 0; i < 7; i++) encoder(M1b, M1b, P_STK, 2, LC, cmask, 5);
  hipMemcpyAsync(M2b, M1b, sizeof(float) * NBL, hipMemcpyDeviceToDevice, stream);
  for (int i = 0; i < 7; i++) encoder(M2b, M2b, P_STK, 2, LC, cmask, 5);
  hipMemcpyAsync(M3b, M2b, sizeof(float) * NBL, hipMemcpyDeviceToDevice, stream);
  for (int i = 0; i < 7; i++) encoder(M3b, M3b, P_STK, 2, LC, cmask, 5);

  // ---- pointer ----
  dot2_k<<<grid1((size_t)Bn * LC), 256, 0, stream>>>(M1b, M2b, in(P_PTR_W1), Y1v, Bn * LC, DM);
  dot2_k<<<grid1((size_t)Bn * LC), 256, 0, stream>>>(M1b, M3b, in(P_PTR_W2), Y2v, Bn * LC, DM);
  float* outp = (float*)d_out;
  gemmS(Y1v, in(P_PTR_L1W), outp, in(P_PTR_L1B), Bn, LC, LC, LC, LC, LC);
  gemmS(Y2v, in(P_PTR_L2W), outp + (size_t)Bn * LC, in(P_PTR_L2B), Bn, LC, LC, LC, LC, LC);
}
